// GIKT_24240795419069
// MI455X (gfx1250) — compile-verified, matrix operand order checked by reading, not run on hardware
//
#include <hip/hip_runtime.h>

#define BSZ   128
#define SEQL  128
#define DIM   128
#define QN    4
#define CN    10
#define NQC   4
#define RANKK 10
#define MB    16
#define NWG   (BSZ / MB)
#define NTHR  256
#define NWAVE (NTHR / 32)
#define NEGF  (-1000000000.0f)

#define LB0   136    // bf16 row pitch (ushorts) for 128-col buffers
#define LB1   264    // bf16 row pitch (ushorts) for 256-col x buffer

typedef __attribute__((ext_vector_type(16))) __bf16 v16bf;
typedef __attribute__((ext_vector_type(8)))  float  v8f;

union BV { v16bf v; unsigned short u[16]; };

__device__ __forceinline__ unsigned short f2bf(float f) {
  unsigned int x = __float_as_uint(f);
  x += 0x7FFFu + ((x >> 16) & 1u);          // round-to-nearest-even
  return (unsigned short)(x >> 16);
}

__device__ __forceinline__ float bf2f(unsigned short b) {
  return __uint_as_float((unsigned int)b << 16);
}

__device__ __forceinline__ float sigf(float x) { return 1.0f / (1.0f + __expf(-x)); }

__device__ __forceinline__ float fast_tanh(float x) {
#if __has_builtin(__builtin_amdgcn_tanhf)
  return __builtin_amdgcn_tanhf(x);          // native V_TANH_F32 on gfx1250
#else
  return tanhf(x);
#endif
}

__device__ __forceinline__ float dotD(const float* a, const float* b) {
  float s = 0.f;
#pragma unroll 8
  for (int d = 0; d < DIM; ++d) s += a[d] * b[d];
  return s;
}

__device__ __forceinline__ float dotBF(const unsigned short* a, const float* b) {
  float s = 0.f;
#pragma unroll 8
  for (int d = 0; d < DIM; ++d) s += bf2f(a[d]) * b[d];
  return s;
}

// ---- WMMA B-operand pre-swizzle: weight W is (N x K) row-major f32; emit
// 1KB tiles (kt,nt): lane l holds 16 bf16, n = nt*16 + (l&15),
// k = kt*32 + (l>>4)*8 + (e&7) + (e>=8 ? 16 : 0)   (mirrors loadA layout)
__device__ void swizzleWeight(const float* W, int N, int K, unsigned short* dst, int tid) {
  int TK = K >> 5, TN = N >> 4;
  int total = TN * TK * 512;
  for (int idx = tid; idx < total; idx += NTHR) {
    int t  = idx >> 9;
    int r  = idx & 511;
    int l  = r >> 4, e = r & 15;
    int nt = t / TK, kt = t % TK;
    int n  = (nt << 4) + (l & 15);
    int k  = (kt << 5) + ((l >> 4) << 3) + (e & 7) + ((e >> 3) << 4);
    dst[idx] = f2bf(W[(size_t)n * K + k]);
  }
}

// A fragment (16x32 bf16) straight from a bf16 LDS buffer: two 16B loads, no VALU
__device__ __forceinline__ v16bf loadA(const unsigned short* A, int lda, int m0, int k0, int lane) {
  const unsigned short* p = A + (size_t)(m0 + (lane & 15)) * lda + k0 + ((lane >> 4) << 3);
  BV r;
  *reinterpret_cast<uint4*>(&r.u[0]) = *reinterpret_cast<const uint4*>(p);
  *reinterpret_cast<uint4*>(&r.u[8]) = *reinterpret_cast<const uint4*>(p + 16);
  return r.v;
}

// OUT[M x N] = act( A[M x K] @ W^T + bias ); A is bf16-staged in LDS.
// TK (= K/32), ACT, WB are compile-time; K-loop fully unrolls.
// WB=1 additionally writes a bf16 mirror of the output (for chained GEMMs).
template <int TK, int ACT, int WB>
__device__ __forceinline__ void gemm(const unsigned short* A, int lda, int M,
                                     const unsigned short* Bsw, int N, const float* bias,
                                     float* OUT, int ldo,
                                     unsigned short* OUTB, int ldob, int tid) {
  const int wave = tid >> 5, lane = tid & 31;
  const int TM = M >> 4, TN = N >> 4;
  for (int tt = wave; tt < TM * TN; tt += NWAVE) {
    int mt = tt / TN, nt = tt % TN;
    const unsigned short* bt = Bsw + (size_t)nt * TK * 512;
    v8f c = {0.f, 0.f, 0.f, 0.f, 0.f, 0.f, 0.f, 0.f};
#pragma unroll
    for (int kt = 0; kt < TK; ++kt) {
      __builtin_prefetch(bt + (kt + 1) * 512 + lane * 16, 0, 0);
      v16bf a = loadA(A, lda, mt << 4, kt << 5, lane);
      v16bf b = ((const v16bf*)(bt + kt * 512))[lane];
      c = __builtin_amdgcn_wmma_f32_16x16x32_bf16(false, a, false, b, (short)0, c, false, false);
    }
    int n    = (nt << 4) + (lane & 15);
    float bn = bias[n];
    int mb   = (mt << 4) + ((lane >> 4) << 3);
#pragma unroll
    for (int i = 0; i < 8; ++i) {
      float v = c[i] + bn;
      if (ACT) v = fast_tanh(v);
      OUT[(size_t)(mb + i) * ldo + n] = v;
      if (WB) OUTB[(size_t)(mb + i) * ldob + n] = f2bf(v);
    }
  }
}

struct GParams {
  const int *qseq, *cseq, *mseq, *qnbr, *cnbr, *qct;
  const float *Eq, *Ec, *Er;
  const float *W_ih1, *W_hh1, *b_ih1, *b_hh1;
  const float *W_ih2, *W_hh2, *b_ih2, *b_hh2;
  const float *W_agg, *b_agg, *W_lst, *b_lst;
  const float *Wq, *bq, *Wk, *bk, *Ww, *bw;
  const float *h1i, *h2i;
  float *y;
  unsigned short *wz;   // pre-swizzled bf16 weight tiles (640 KB)
  float *hist;          // state_hist (BS, SEQ, D) f32 (8 MB)
};

// tile offsets (ushort units) within wz
#define OFF_IH1 0
#define OFF_HH1 (192 * 512)
#define OFF_IH2 (288 * 512)
#define OFF_HH2 (384 * 512)
#define OFF_AG0 (480 * 512)
#define OFF_AG1 (512 * 512)
#define OFF_LST (544 * 512)
#define OFF_WQ  (576 * 512)
#define OFF_WK  (608 * 512)

__global__ __launch_bounds__(NTHR, 1)
void gikt_cdna5_kernel(GParams P) {
  const int tid = threadIdx.x;
  const int b0  = blockIdx.x * MB;

  __shared__ __align__(16) float sm_h1[MB * 132];
  __shared__ __align__(16) float sm_h2[MB * 132];
  __shared__ __align__(16) float sm_g2[MB * 132];
  __shared__ __align__(16) float sm_e0[MB * 132];
  __shared__ __align__(16) float sP0[MB * 132];       // emb0 / agg / eqn
  __shared__ __align__(16) float sP1[64 * 132];       // emb1,e1 / gi / Kp
  __shared__ __align__(16) float sP2[64 * 132];       // emb2mean / gh / Qp
  __shared__ __align__(16) float sP3[MB * SEQL];      // scores
  __shared__ __align__(16) unsigned short sB0[64 * LB0];  // bf16 A: s1 / e0' / qc / hist
  __shared__ __align__(16) unsigned short sB1[MB * LB1];  // bf16 A: s0,s0' / x
  __shared__ __align__(16) unsigned short sBh1[MB * LB0]; // bf16 mirror of h1
  __shared__ __align__(16) unsigned short sBh2[MB * LB0]; // bf16 mirror of h2
  __shared__ int   si_q[MB], si_r[MB], si_m[MB], si_qn[MB];
  __shared__ int   si_n1[MB * QN];
  __shared__ int   si_n2[MB * QN * CN];
  __shared__ int   si_qc[MB * NQC];
  __shared__ int   si_top[MB * RANKK];
  __shared__ int   si_msk[MB * RANKK];
  __shared__ float sf_hh[4 * 16], sf_cc[4 * 16], sf_lg[4 * 55];

  // ---- one-time: swizzle all weights to bf16 B-tiles (redundant, identical across WGs)
  swizzleWeight(P.W_ih1, 384, 256, P.wz + OFF_IH1, tid);
  swizzleWeight(P.W_hh1, 384, 128, P.wz + OFF_HH1, tid);
  swizzleWeight(P.W_ih2, 384, 128, P.wz + OFF_IH2, tid);
  swizzleWeight(P.W_hh2, 384, 128, P.wz + OFF_HH2, tid);
  swizzleWeight(P.W_agg,             128, 128, P.wz + OFF_AG0, tid);
  swizzleWeight(P.W_agg + DIM * DIM, 128, 128, P.wz + OFF_AG1, tid);
  swizzleWeight(P.W_lst, 128, 128, P.wz + OFF_LST, tid);
  swizzleWeight(P.Wq,    128, 128, P.wz + OFF_WQ,  tid);
  swizzleWeight(P.Wk,    128, 128, P.wz + OFF_WK,  tid);

  {
    float* hb = P.hist + (size_t)b0 * SEQL * DIM;
    for (int i = tid; i < MB * SEQL * DIM; i += NTHR) hb[i] = 0.f;
    for (int i = tid; i < MB * SEQL; i += NTHR) P.y[(size_t)b0 * SEQL + i] = 0.f;
    for (int i = tid; i < MB * DIM; i += NTHR) {
      int m = i >> 7, d = i & 127;
      float a = P.h1i[(size_t)(b0 + m) * DIM + d];
      float b = P.h2i[(size_t)(b0 + m) * DIM + d];
      sm_h1[m * 132 + d] = a;  sBh1[m * LB0 + d] = f2bf(a);
      sm_h2[m * 132 + d] = b;  sBh2[m * LB0 + d] = f2bf(b);
    }
  }
  __syncthreads();

  for (int t = 0; t < SEQL - 1; ++t) {
    // ---------------- gathers ----------------
    if (tid < MB) {
      int b = b0 + tid;
      si_q[tid]  = P.qseq[(size_t)b * SEQL + t];
      si_r[tid]  = P.cseq[(size_t)b * SEQL + t];
      si_m[tid]  = P.mseq[(size_t)b * SEQL + t];
      si_qn[tid] = P.qseq[(size_t)b * SEQL + t + 1];
    }
    __syncthreads();
    if (tid < MB * QN)  si_n1[tid] = P.qnbr[(size_t)si_q[tid >> 2] * QN + (tid & 3)];
    if (tid < MB * NQC) si_qc[tid] = P.qct[(size_t)si_qn[tid >> 2] * NQC + (tid & 3)];
    __syncthreads();
    for (int i = tid; i < MB * QN * CN; i += NTHR)
      si_n2[i] = P.cnbr[(size_t)si_n1[i / CN] * CN + (i % CN)];
    __syncthreads();

    // ---------------- graph aggregation ----------------
    for (int i = tid; i < MB * DIM; i += NTHR) {
      int m = i >> 7, d = i & 127;
      sP0[m * 132 + d] = P.Eq[(size_t)si_q[m] * DIM + d];            // emb0
    }
    for (int i = tid; i < MB * QN * DIM; i += NTHR) {
      int rm = i >> 7, d = i & 127;
      sP1[rm * 132 + d] = P.Ec[(size_t)si_n1[rm] * DIM + d];         // emb1
      float acc = 0.f;
      const int* nn = si_n2 + rm * CN;
      for (int c = 0; c < CN; ++c) acc += P.Eq[(size_t)nn[c] * DIM + d];
      sP2[rm * 132 + d] = acc * 0.1f;                                // mean(emb2)
    }
    __syncthreads();
    for (int i = tid; i < MB * DIM; i += NTHR) {                     // s0 = mean(emb1)+emb0 (bf16)
      int m = i >> 7, d = i & 127;
      float s = 0.f;
      for (int j = 0; j < QN; ++j) s += sP1[(m * QN + j) * 132 + d];
      sB1[m * LB1 + d] = f2bf(s * 0.25f + sP0[m * 132 + d]);
    }
    for (int i = tid; i < MB * QN * DIM; i += NTHR) {                // s1 = mean(emb2)+emb1 (bf16)
      int rm = i >> 7, d = i & 127;
      sB0[rm * LB0 + d] = f2bf(sP2[rm * 132 + d] + sP1[rm * 132 + d]);
    }
    __syncthreads();
    gemm<4, 1, 0>(sB1, LB1, MB,      P.wz + OFF_AG0, DIM, P.b_agg,       sm_e0, 132, (unsigned short*)nullptr, 0, tid);
    gemm<4, 1, 0>(sB0, LB0, MB * QN, P.wz + OFF_AG1, DIM, P.b_agg + DIM, sP1,   132, (unsigned short*)nullptr, 0, tid);
    __syncthreads();
    for (int i = tid; i < MB * DIM; i += NTHR) {                     // s0' = mean(e1)+e0 (bf16)
      int m = i >> 7, d = i & 127;
      float s = 0.f;
      for (int j = 0; j < QN; ++j) s += sP1[(m * QN + j) * 132 + d];
      sB1[m * LB1 + d] = f2bf(s * 0.25f + sm_e0[m * 132 + d]);
    }
    __syncthreads();
    gemm<4, 1, 1>(sB1, LB1, MB, P.wz + OFF_AG0, DIM, P.b_agg, sm_e0, 132, sB0, LB0, tid);  // e0' (+bf16)
    __syncthreads();
    gemm<4, 1, 0>(sB0, LB0, MB, P.wz + OFF_LST, DIM, P.b_lst, sP0, 132, (unsigned short*)nullptr, 0, tid);  // agg
    __syncthreads();

    // x = [emb_q, Er[r]]  (bf16, pitch LB1)
    for (int i = tid; i < MB * 256; i += NTHR) {
      int m = i >> 8, d = i & 255;
      float v;
      if (d < DIM) v = si_m[m] ? sP0[m * 132 + d] : P.Eq[(size_t)si_q[m] * DIM + d];
      else         v = P.Er[(size_t)si_r[m] * DIM + (d - 128)];
      sB1[m * LB1 + d] = f2bf(v);
    }
    __syncthreads();

    // ---------------- GRU1 ----------------
    gemm<8, 0, 0>(sB1,  LB1, MB, P.wz + OFF_IH1, 384, P.b_ih1, sP1, 388, (unsigned short*)nullptr, 0, tid);
    gemm<4, 0, 0>(sBh1, LB0, MB, P.wz + OFF_HH1, 384, P.b_hh1, sP2, 388, (unsigned short*)nullptr, 0, tid);
    __syncthreads();
    for (int i = tid; i < MB * DIM; i += NTHR) {
      int m = i >> 7, d = i & 127;
      float r = sigf(sP1[m * 388 + d]       + sP2[m * 388 + d]);
      float z = sigf(sP1[m * 388 + 128 + d] + sP2[m * 388 + 128 + d]);
      float n = fast_tanh(sP1[m * 388 + 256 + d] + r * sP2[m * 388 + 256 + d]);
      float h = (1.f - z) * n + z * sm_h1[m * 132 + d];
      sm_h1[m * 132 + d] = h;
      sBh1[m * LB0 + d]  = f2bf(h);
    }
    __syncthreads();

    // ---------------- GRU2 -> g2 ----------------
    gemm<4, 0, 0>(sBh1, LB0, MB, P.wz + OFF_IH2, 384, P.b_ih2, sP1, 388, (unsigned short*)nullptr, 0, tid);
    gemm<4, 0, 0>(sBh2, LB0, MB, P.wz + OFF_HH2, 384, P.b_hh2, sP2, 388, (unsigned short*)nullptr, 0, tid);
    __syncthreads();
    for (int i = tid; i < MB * DIM; i += NTHR) {
      int m = i >> 7, d = i & 127;
      float r = sigf(sP1[m * 388 + d]       + sP2[m * 388 + d]);
      float z = sigf(sP1[m * 388 + 128 + d] + sP2[m * 388 + 128 + d]);
      float n = fast_tanh(sP1[m * 388 + 256 + d] + r * sP2[m * 388 + 256 + d]);
      sm_g2[m * 132 + d] = (1.f - z) * n + z * sm_h2[m * 132 + d];
    }
    __syncthreads();

    // ---------------- history scores + top-k ----------------
    for (int i = tid; i < MB * DIM; i += NTHR) {
      int m = i >> 7, d = i & 127;
      sP0[m * 132 + d] = P.Eq[(size_t)si_qn[m] * DIM + d];           // eqn
    }
    __syncthreads();
    for (int i = tid; i < MB * SEQL; i += NTHR) {
      int m = i >> 7, s = i & 127;
      float v = NEGF;
      if (s < t) {
        const float* row = P.Eq + (size_t)P.qseq[(size_t)(b0 + m) * SEQL + s] * DIM;
        v = dotD(row, sP0 + m * 132);
      }
      sP3[i] = v;
    }
    __syncthreads();
    if (tid < MB) {
      float* sc = sP3 + tid * SEQL;
      for (int k = 0; k < RANKK; ++k) {
        float best = sc[0]; int bi = 0;
        for (int s = 1; s < SEQL; ++s) { float x = sc[s]; if (x > best) { best = x; bi = s; } }
        si_top[tid * RANKK + k] = bi;
        si_msk[tid * RANKK + k] = (best > NEGF * 0.5f) ? 1 : 0;
        sc[bi] = NEGF * 4.0f;
      }
    }
    __syncthreads();

    // ---------------- attention readout, 4 batch rows per pass ----------------
    const int yidx = (t == 0) ? 0 : (t + 1);
    for (int mc = 0; mc < MB; mc += 4) {
      // stage qc (bf16, pad to 16 rows per m)
      for (int i = tid; i < 64 * DIM; i += NTHR) {
        int rr = i >> 7, d = i & 127;
        int mi = rr >> 4, j = rr & 15, m = mc + mi;
        float v = 0.f;
        if (j == 0)        v = P.Eq[(size_t)si_qn[m] * DIM + d];
        else if (j <= NQC) v = P.Ec[(size_t)si_qc[m * NQC + (j - 1)] * DIM + d];
        sB0[rr * LB0 + d] = f2bf(v);
      }
      __syncthreads();
      if (tid < 4 * (NQC + 1)) {
        int mi = tid / 5, j = tid % 5;
        sf_cc[mi * 16 + j] = dotBF(sB0 + (mi * 16 + j) * LB0, P.Ww + DIM);
      }
      __syncthreads();
      gemm<4, 0, 0>(sB0, LB0, 64, P.wz + OFF_WK, DIM, P.bk, sP1, 132, (unsigned short*)nullptr, 0, tid);  // Kp
      __syncthreads();
      // stage hist (bf16: g2 + top-k selected state rows, pad to 16)
      for (int i = tid; i < 64 * DIM; i += NTHR) {
        int rr = i >> 7, d = i & 127;
        int mi = rr >> 4, h = rr & 15, m = mc + mi;
        float v = 0.f;
        if (h == 0)          v = sm_g2[m * 132 + d];
        else if (h <= RANKK) v = P.hist[((size_t)(b0 + m) * SEQL + si_top[m * RANKK + h - 1]) * DIM + d];
        sB0[rr * LB0 + d] = f2bf(v);
      }
      __syncthreads();
      if (tid < 4 * (RANKK + 1)) {
        int mi = tid / 11, h = tid % 11;
        sf_hh[mi * 16 + h] = dotBF(sB0 + (mi * 16 + h) * LB0, P.Ww);
      }
      __syncthreads();
      gemm<4, 0, 0>(sB0, LB0, 64, P.wz + OFF_WQ, DIM, P.bq, sP2, 132, (unsigned short*)nullptr, 0, tid);  // Qp
      __syncthreads();
      for (int i = tid; i < 4 * 55; i += NTHR) {
        int mi = i / 55, r = i % 55;
        int hi = r / 5, j = r % 5, m = mc + mi;
        float l;
        if (hi > 0 && !si_msk[m * RANKK + hi - 1]) l = NEGF;
        else l = dotD(sP2 + (mi * 16 + hi) * 132, sP1 + (mi * 16 + j) * 132);
        sf_lg[i] = l;
      }
      __syncthreads();
      if (tid < 4) {
        int mi = tid, m = mc + mi;
        const float bw0 = P.bw[0];
        float mx = -3.4e38f;
        for (int r = 0; r < 55; ++r) mx = fmaxf(mx, sf_lg[mi * 55 + r]);
        float se = 0.f;
        for (int r = 0; r < 55; ++r) { float e = __expf(sf_lg[mi * 55 + r] - mx); sf_lg[mi * 55 + r] = e; se += e; }
        float p = 0.f;
        for (int r = 0; r < 55; ++r) {
          int hi = r / 5, j = r % 5;
          float v = sigf(sf_hh[mi * 16 + hi] + sf_cc[mi * 16 + j] + bw0);
          p += (sf_lg[mi * 55 + r] / se) * v;
        }
        P.y[(size_t)(b0 + m) * SEQL + yidx] = p;
      }
      __syncthreads();
    }

    // ---------------- end-of-step state update ----------------
    if (t > 0) {
      for (int i = tid; i < MB * DIM; i += NTHR) {
        int m = i >> 7, d = i & 127;
        float v = sm_g2[m * 132 + d];
        sm_h2[m * 132 + d] = v;
        sBh2[m * LB0 + d]  = f2bf(v);
        P.hist[((size_t)(b0 + m) * SEQL + t) * DIM + d] = v;
      }
    }
    __syncthreads();
  }
}

extern "C" void kernel_launch(void* const* d_in, const int* in_sizes, int n_in,
                              void* d_out, int out_size, void* d_ws, size_t ws_size,
                              hipStream_t stream) {
  (void)in_sizes; (void)n_in; (void)out_size; (void)ws_size;
  GParams P;
  P.qseq  = (const int*)d_in[0];
  P.cseq  = (const int*)d_in[1];
  P.mseq  = (const int*)d_in[2];
  P.qnbr  = (const int*)d_in[3];
  P.cnbr  = (const int*)d_in[4];
  P.qct   = (const int*)d_in[5];
  P.Eq    = (const float*)d_in[6];
  P.Ec    = (const float*)d_in[7];
  P.Er    = (const float*)d_in[8];
  P.W_ih1 = (const float*)d_in[9];
  P.W_hh1 = (const float*)d_in[10];
  P.b_ih1 = (const float*)d_in[11];
  P.b_hh1 = (const float*)d_in[12];
  P.W_ih2 = (const float*)d_in[13];
  P.W_hh2 = (const float*)d_in[14];
  P.b_ih2 = (const float*)d_in[15];
  P.b_hh2 = (const float*)d_in[16];
  P.W_agg = (const float*)d_in[17];
  P.b_agg = (const float*)d_in[18];
  P.W_lst = (const float*)d_in[19];
  P.b_lst = (const float*)d_in[20];
  P.Wq    = (const float*)d_in[21];
  P.bq    = (const float*)d_in[22];
  P.Wk    = (const float*)d_in[23];
  P.bk    = (const float*)d_in[24];
  P.Ww    = (const float*)d_in[25];
  P.bw    = (const float*)d_in[26];
  P.h1i   = (const float*)d_in[27];
  P.h2i   = (const float*)d_in[28];
  P.y     = (float*)d_out;
  P.wz    = (unsigned short*)d_ws;                       // 640 KB swizzled weights
  P.hist  = (float*)((char*)d_ws + 640 * 1024);          // 8 MB state_hist
  hipLaunchKernelGGL(gikt_cdna5_kernel, dim3(NWG), dim3(NTHR), 0, stream, P);
}
